// CompressedLinear_7988639170887
// MI455X (gfx1250) — compile-verified, matrix operand order checked by reading, not run on hardware
//
#include <hip/hip_runtime.h>
#include <hip/hip_fp16.h>

typedef __attribute__((ext_vector_type(16))) _Float16 v16h;
typedef __attribute__((ext_vector_type(8)))  _Float16 v8h;
typedef __attribute__((ext_vector_type(8)))  float    v8f;
typedef __attribute__((ext_vector_type(4)))  float    v4f;

#define K_DIM 4096
#define N_DIM 16384

// ---------------------------------------------------------------------------
// Pre-pass 1: W int8 -> f16 (exact). 16 elements per thread.
// ---------------------------------------------------------------------------
__global__ __launch_bounds__(256) void convert_w_kernel(
    const signed char* __restrict__ w, _Float16* __restrict__ wh)
{
  const size_t gid = (size_t)blockIdx.x * 256 + threadIdx.x;   // 16-byte chunk id
  int4 q = ((const int4*)w)[gid];
  int qs[4] = {q.x, q.y, q.z, q.w};
  v16h h;
#pragma unroll
  for (int b = 0; b < 4; ++b) {
    h[4 * b + 0] = (_Float16)((signed char)(qs[b]));
    h[4 * b + 1] = (_Float16)((signed char)(qs[b] >> 8));
    h[4 * b + 2] = (_Float16)((signed char)(qs[b] >> 16));
    h[4 * b + 3] = (_Float16)((signed char)(qs[b] >> 24));
  }
  ((v16h*)wh)[gid] = h;
}

// ---------------------------------------------------------------------------
// Pre-pass 2: x fp32 -> f16. 8 elements per thread.
// ---------------------------------------------------------------------------
__global__ __launch_bounds__(256) void convert_x_kernel(
    const float* __restrict__ x, _Float16* __restrict__ xh)
{
  const size_t gid = (size_t)blockIdx.x * 256 + threadIdx.x;   // 8-float chunk id
  const v4f* px = (const v4f*)x;
  v4f f0 = px[2 * gid], f1 = px[2 * gid + 1];
  v8h h;
#pragma unroll
  for (int j = 0; j < 4; ++j) {
    h[j]     = (_Float16)f0[j];
    h[4 + j] = (_Float16)f1[j];
  }
  ((v8h*)xh)[gid] = h;
}

// ---------------------------------------------------------------------------
// Main GEMM: y[M,N] = (xh[M,K] @ wh[N,K]^T) * scale + bias[N]
// Wave tile 64(M) x 64(N): 4 A frags x 4 B frags -> 16 WMMAs per 16 b128
// loads per 32-K step (A and B each reused 4x). Pure loads + v_wmma in the
// hot loop. Block: 8 waves (2M x 4N) = 128M x 256N.
// ---------------------------------------------------------------------------
__device__ inline v16h ldfrag(const _Float16* p, int kOff0, int kOff1) {
  // ISA 16-bit fragment layout: halves 0..7 <- K=[kOff0,kOff0+8),
  // halves 8..15 <- K=[kOff1,kOff1+8). Each chunk is one 16B load.
  v8h c0 = *(const v8h*)(p + kOff0);
  v8h c1 = *(const v8h*)(p + kOff1);
  return __builtin_shufflevector(c0, c1, 0, 1, 2, 3, 4, 5, 6, 7,
                                 8, 9, 10, 11, 12, 13, 14, 15);
}

__global__ __launch_bounds__(256) void gemm_f16_wmma(
    const _Float16* __restrict__ xh,
    const _Float16* __restrict__ wh,
    const float* __restrict__ scale_p,
    const __half* __restrict__ bias,
    float* __restrict__ out)
{
  const int lane = threadIdx.x & 31;
  const int wave = threadIdx.x >> 5;
  const int l16  = lane & 15;
  const int hi   = lane >> 4;

  const int wm = wave & 1;             // wave row in block (M): 0..1
  const int wn = wave >> 1;            // wave col in block (N): 0..3

  const int mTile = blockIdx.y * 128 + wm * 64;
  const int nTile = blockIdx.x * 256 + wn * 64;

  const _Float16* aP[4];
  const _Float16* bP[4];
#pragma unroll
  for (int i = 0; i < 4; ++i) {
    aP[i] = xh + (size_t)(mTile + 16 * i + l16) * K_DIM;
    bP[i] = wh + (size_t)(nTile + 16 * i + l16) * K_DIM;
  }

  const int kOff0 = hi * 8;
  const int kOff1 = 16 + hi * 8;

  v8f acc[4][4];
#pragma unroll
  for (int i = 0; i < 4; ++i)
#pragma unroll
    for (int j = 0; j < 4; ++j)
      acc[i][j] = (v8f){};

  for (int k = 0; k < K_DIM; k += 32) {
    v16h A[4], B[4];
#pragma unroll
    for (int i = 0; i < 4; ++i) A[i] = ldfrag(aP[i] + k, kOff0, kOff1);
#pragma unroll
    for (int j = 0; j < 4; ++j) B[j] = ldfrag(bP[j] + k, kOff0, kOff1);

#pragma unroll
    for (int i = 0; i < 4; ++i)
#pragma unroll
      for (int j = 0; j < 4; ++j)
        acc[i][j] = __builtin_amdgcn_wmma_f32_16x16x32_f16(
            false, A[i], false, B[j], (short)0, acc[i][j], false, false);
  }

  const float scale = scale_p[0];
  float bs[4];
#pragma unroll
  for (int j = 0; j < 4; ++j)
    bs[j] = __half2float(bias[nTile + 16 * j + l16]);

  // C/D layout: VGPR r -> M = r + 8*hi, N = l16.
#pragma unroll
  for (int i = 0; i < 4; ++i) {
#pragma unroll
    for (int r = 0; r < 8; ++r) {
      const size_t mrow = (size_t)(mTile + 16 * i + r + hi * 8) * N_DIM;
#pragma unroll
      for (int j = 0; j < 4; ++j)
        out[mrow + nTile + 16 * j + l16] = acc[i][j][r] * scale + bs[j];
    }
  }
}

// ---------------------------------------------------------------------------
// Fallback (only if d_ws is too small for pre-converted operands):
// fused-conversion kernel, wave tile 32x32.
// ---------------------------------------------------------------------------
__global__ __launch_bounds__(256) void compressed_linear_wmma_fused(
    const float* __restrict__ x,
    const signed char* __restrict__ w,
    const float* __restrict__ scale_p,
    const __half* __restrict__ bias,
    float* __restrict__ out)
{
  const int lane = threadIdx.x & 31;
  const int wave = threadIdx.x >> 5;
  const int l16  = lane & 15;
  const int hi   = lane >> 4;

  const int wm = wave & 3;
  const int wn = wave >> 2;

  const int mTile = blockIdx.y * 128 + wm * 32;
  const int nTile = blockIdx.x * 64  + wn * 32;

  const float*       aRow0 = x + (size_t)(mTile +      l16) * K_DIM;
  const float*       aRow1 = x + (size_t)(mTile + 16 + l16) * K_DIM;
  const signed char* bRow0 = w + (size_t)(nTile +      l16) * K_DIM;
  const signed char* bRow1 = w + (size_t)(nTile + 16 + l16) * K_DIM;

  const int kOff0 = hi * 8;
  const int kOff1 = 16 + hi * 8;

  v8f acc00 = {}, acc01 = {}, acc10 = {}, acc11 = {};

  for (int k = 0; k < K_DIM; k += 32) {
    v16h a0, a1;
    {
      const v4f* p0 = (const v4f*)(aRow0 + k + kOff0);
      const v4f* p1 = (const v4f*)(aRow0 + k + kOff1);
      v4f f0 = p0[0], f1 = p0[1], f2 = p1[0], f3 = p1[1];
#pragma unroll
      for (int j = 0; j < 4; ++j) {
        a0[j] = (_Float16)f0[j]; a0[4 + j] = (_Float16)f1[j];
        a0[8 + j] = (_Float16)f2[j]; a0[12 + j] = (_Float16)f3[j];
      }
    }
    {
      const v4f* p0 = (const v4f*)(aRow1 + k + kOff0);
      const v4f* p1 = (const v4f*)(aRow1 + k + kOff1);
      v4f f0 = p0[0], f1 = p0[1], f2 = p1[0], f3 = p1[1];
#pragma unroll
      for (int j = 0; j < 4; ++j) {
        a1[j] = (_Float16)f0[j]; a1[4 + j] = (_Float16)f1[j];
        a1[8 + j] = (_Float16)f2[j]; a1[12 + j] = (_Float16)f3[j];
      }
    }
    v16h b0, b1;
    {
      int2 q0 = *(const int2*)(bRow0 + k + kOff0);
      int2 q1 = *(const int2*)(bRow0 + k + kOff1);
#pragma unroll
      for (int j = 0; j < 4; ++j) {
        b0[j]      = (_Float16)((signed char)(q0.x >> (8 * j)));
        b0[4 + j]  = (_Float16)((signed char)(q0.y >> (8 * j)));
        b0[8 + j]  = (_Float16)((signed char)(q1.x >> (8 * j)));
        b0[12 + j] = (_Float16)((signed char)(q1.y >> (8 * j)));
      }
    }
    {
      int2 q0 = *(const int2*)(bRow1 + k + kOff0);
      int2 q1 = *(const int2*)(bRow1 + k + kOff1);
#pragma unroll
      for (int j = 0; j < 4; ++j) {
        b1[j]      = (_Float16)((signed char)(q0.x >> (8 * j)));
        b1[4 + j]  = (_Float16)((signed char)(q0.y >> (8 * j)));
        b1[8 + j]  = (_Float16)((signed char)(q1.x >> (8 * j)));
        b1[12 + j] = (_Float16)((signed char)(q1.y >> (8 * j)));
      }
    }
    acc00 = __builtin_amdgcn_wmma_f32_16x16x32_f16(false, a0, false, b0,
                                                   (short)0, acc00, false, false);
    acc01 = __builtin_amdgcn_wmma_f32_16x16x32_f16(false, a0, false, b1,
                                                   (short)0, acc01, false, false);
    acc10 = __builtin_amdgcn_wmma_f32_16x16x32_f16(false, a1, false, b0,
                                                   (short)0, acc10, false, false);
    acc11 = __builtin_amdgcn_wmma_f32_16x16x32_f16(false, a1, false, b1,
                                                   (short)0, acc11, false, false);
  }

  const float scale = scale_p[0];
  const float bias0 = __half2float(bias[nTile +      l16]);
  const float bias1 = __half2float(bias[nTile + 16 + l16]);
#pragma unroll
  for (int r = 0; r < 8; ++r) {
    const int m0 = mTile +      r + hi * 8;
    const int m1 = mTile + 16 + r + hi * 8;
    out[(size_t)m0 * N_DIM + nTile +      l16] = acc00[r] * scale + bias0;
    out[(size_t)m0 * N_DIM + nTile + 16 + l16] = acc01[r] * scale + bias1;
    out[(size_t)m1 * N_DIM + nTile +      l16] = acc10[r] * scale + bias0;
    out[(size_t)m1 * N_DIM + nTile + 16 + l16] = acc11[r] * scale + bias1;
  }
}

extern "C" void kernel_launch(void* const* d_in, const int* in_sizes, int n_in,
                              void* d_out, int out_size, void* d_ws, size_t ws_size,
                              hipStream_t stream) {
  const float*       x     = (const float*)d_in[0];
  const signed char* w     = (const signed char*)d_in[1];
  const float*       scale = (const float*)d_in[2];
  const __half*      bias  = (const __half*)d_in[3];
  float*             out   = (float*)d_out;

  const int M = in_sizes[0] / K_DIM;                 // 2*2048 = 4096
  const size_t wBytes = (size_t)N_DIM * K_DIM * sizeof(_Float16);   // 128 MB
  const size_t xBytes = (size_t)M * K_DIM * sizeof(_Float16);       //  32 MB

  if (ws_size >= wBytes + xBytes) {
    _Float16* wh = (_Float16*)d_ws;
    _Float16* xh = (_Float16*)((char*)d_ws + wBytes);

    const int wBlocks = (int)(((size_t)N_DIM * K_DIM / 16) / 256);  // 16384
    const int xBlocks = (int)(((size_t)M * K_DIM / 8) / 256);       //  8192
    hipLaunchKernelGGL(convert_w_kernel, dim3(wBlocks), dim3(256), 0, stream, w, wh);
    hipLaunchKernelGGL(convert_x_kernel, dim3(xBlocks), dim3(256), 0, stream, x, xh);

    dim3 grid(N_DIM / 256, M / 128);
    hipLaunchKernelGGL(gemm_f16_wmma, grid, dim3(256), 0, stream,
                       xh, wh, scale, bias, out);
  } else {
    dim3 grid(N_DIM / 64, M / 128);
    hipLaunchKernelGGL(compressed_linear_wmma_fused, grid, dim3(256), 0, stream,
                       x, w, scale, bias, out);
  }
}
